// FastRNNlayer_32873679684250
// MI455X (gfx1250) — matrix operand order repaired and checked by reading, not compiled, not run on hardware
//
#include <hip/hip_runtime.h>
#include <hip/hip_bf16.h>
#include <cstddef>

// Problem constants (from reference)
#define S_LEN 512
#define BSZ   8
#define NH    8
#define HD    64
#define IND   512
#define PROJ  2576              // NH*(5*HD+2)
#define ROWS  (S_LEN*BSZ)       // 4096
#define PROJ_PAD 2624           // 41*64 : zero-padded B rows for GEMM1

typedef __attribute__((ext_vector_type(16))) _Float16 v16h;
typedef __attribute__((ext_vector_type(8)))  _Float16 v8h;
typedef __attribute__((ext_vector_type(8)))  float    v8f;

// LDS B-tile: 64 rows x 32 halfs of K, row stride padded to 40 halfs (80 B)
// -> 16-B aligned b128 accesses, banks spread (gcd(20,64)=4 cycle).
#define BT_STRIDE 40

// ---------------------------------------------------------------------------
// LayerNorm (biased var, eps=1e-5) -> f16 normed activations. 1 wave per row.
// ---------------------------------------------------------------------------
__global__ __launch_bounds__(128)
void ln_kernel(const float* __restrict__ x, const float* __restrict__ gamma,
               const float* __restrict__ beta, _Float16* __restrict__ out, int rows)
{
    int row  = blockIdx.x * 4 + (threadIdx.x >> 5);
    int lane = threadIdx.x & 31;
    if (row >= rows) return;
    const float* xr = x + (size_t)row * IND;
    float v[16];
    float s = 0.f, s2 = 0.f;
#pragma unroll
    for (int e = 0; e < 16; ++e) {
        v[e] = xr[lane + e * 32];
        s  += v[e];
        s2 += v[e] * v[e];
    }
#pragma unroll
    for (int m = 16; m; m >>= 1) {
        s  += __shfl_xor(s,  m, 32);
        s2 += __shfl_xor(s2, m, 32);
    }
    float mu   = s  * (1.f / IND);
    float var  = s2 * (1.f / IND) - mu * mu;
    float rstd = rsqrtf(var + 1e-5f);
    _Float16* orow = out + (size_t)row * IND;
#pragma unroll
    for (int e = 0; e < 16; ++e) {
        int c = lane + e * 32;
        orow[c] = (_Float16)((v[e] - mu) * rstd * gamma[c] + beta[c]);
    }
}

// ---------------------------------------------------------------------------
// f32 (rows x cols) -> f16 (padRows x cols), zero-filled beyond rows.
// ---------------------------------------------------------------------------
__global__ __launch_bounds__(256)
void cvt_f16_pad_kernel(const float* __restrict__ in, _Float16* __restrict__ out,
                        int rows, int cols, int padRows)
{
    int i = blockIdx.x * 256 + threadIdx.x;
    int total = padRows * cols;
    if (i >= total) return;
    int r = i / cols;
    out[i] = (r < rows) ? (_Float16)in[i] : (_Float16)0.f;
}

// ---------------------------------------------------------------------------
// NT GEMM via v_wmma_f32_16x16x32_f16:  C[M,N] = A[M,K] * B[N,K]^T (+residual)
// 4 waves / block, each wave a 16-row tile; all waves share one 64-col tile.
// Double-buffered B staging: tile kb+32 copied global->LDS with
// global_load_async_to_lds_b128 (ASYNCcnt) while tile kb feeds the WMMAs;
// A fragments are software-pipelined in registers. B buffer must have
// >= gridDim.y*64 rows (zero-padded). N must be a multiple of 16.
// ---------------------------------------------------------------------------
__global__ __launch_bounds__(128)
void gemm_nt_wmma(const _Float16* __restrict__ A, const _Float16* __restrict__ Bm,
                  const float* __restrict__ residual, float* __restrict__ C,
                  int M, int N, int K)
{
    __shared__ _Float16 btile[2][64 * BT_STRIDE];

    const int tid   = threadIdx.x;
    const int lane  = tid & 31;
    const int wave  = tid >> 5;
    const int m0    = (blockIdx.x * 4 + wave) * 16;  // 16-row tile base
    const int n0    = blockIdx.y * 64;               // 64-col tile base

    const int g  = lane >> 4;    // lane group 0/1
    const int ml = lane & 15;

    const _Float16* Arow = A + (size_t)(m0 + ml) * K;
    const unsigned lbase[2] = {
        (unsigned)(unsigned long long)(&btile[0][0]),
        (unsigned)(unsigned long long)(&btile[1][0])
    };

    // 256 chunks of 16 B per tile; chunk c -> row = c>>2, part = c&3 (8 halfs)
    auto issue_copy = [&](int kb, int ib) {
#pragma unroll
        for (int r = 0; r < 2; ++r) {
            int c    = r * 128 + tid;
            int row  = c >> 2;
            int part = c & 3;
            const _Float16* gptr = Bm + (size_t)(n0 + row) * K + kb + part * 8;
            unsigned laddr = lbase[ib] + (unsigned)(row * (BT_STRIDE * 2) + part * 16);
            asm volatile("global_load_async_to_lds_b128 %0, %1, off"
                         :: "v"(laddr), "v"(gptr) : "memory");
        }
    };

    v8f acc[4];
#pragma unroll
    for (int t = 0; t < 4; ++t)
#pragma unroll
        for (int e = 0; e < 8; ++e) acc[t][e] = 0.f;

    const int T = K / 32;                 // K-steps
    issue_copy(0, 0);                     // prologue: tile 0 -> buf 0

    // software-pipelined A fragment registers
    v8h alo = *(const v8h*)(Arow + 0 + g * 8);
    v8h ahi = *(const v8h*)(Arow + 16 + g * 8);

    for (int it = 0; it < T; ++it) {
        const int ib = it & 1;
        const int kb = it * 32;

        // wait for this tile's own async copies (next tile not yet issued)
        asm volatile("s_wait_asynccnt 0" ::: "memory");
        // all waves: tile it visible, and everyone finished reading buf[1-ib]
        __syncthreads();
        if (it + 1 < T) issue_copy(kb + 32, 1 - ib);  // overlap with compute

        // prefetch next A fragment
        v8h alo_n = alo, ahi_n = ahi;
        if (it + 1 < T) {
            alo_n = *(const v8h*)(Arow + kb + 32 + g * 8);
            ahi_n = *(const v8h*)(Arow + kb + 48 + g * 8);
        }

        // A fragment (16x32 f16, ISA A-layout)
        v16h afrag;
#pragma unroll
        for (int e = 0; e < 8; ++e) { afrag[e] = alo[e]; afrag[e + 8] = ahi[e]; }

        const _Float16* bbuf = &btile[ib][0];
#pragma unroll
        for (int ns = 0; ns < 4; ++ns) {
            // B fragment (32x16): lane = column, K = 8g..8g+7 and 16+8g..16+8g+7 per lane group
            const _Float16* brow = bbuf + (ns * 16 + ml) * BT_STRIDE + g * 8;
            v8h blo = *(const v8h*)(brow);
            v8h bhi = *(const v8h*)(brow + 16);
            v16h bfrag;
#pragma unroll
            for (int e = 0; e < 8; ++e) { bfrag[e] = blo[e]; bfrag[e + 8] = bhi[e]; }

            acc[ns] = __builtin_amdgcn_wmma_f32_16x16x32_f16(
                /*neg_a=*/false, afrag, /*neg_b=*/false, bfrag,
                /*c_mod=*/(short)0, acc[ns], /*reuse_a=*/false, /*reuse_b=*/false);
        }
        alo = alo_n; ahi = ahi_n;
    }

    // C/D layout: VGPR r -> row m0 + r + 8*g, column n = ncol + (lane&15)
#pragma unroll
    for (int ns = 0; ns < 4; ++ns) {
        const int ncol = n0 + ns * 16;
        if (ncol + 15 < N) {            // wave-uniform (N % 16 == 0)
            const int n = ncol + ml;
#pragma unroll
            for (int r = 0; r < 8; ++r) {
                const int m = m0 + r + 8 * g;
                size_t idx = (size_t)m * N + n;
                float vv = acc[ns][r];
                if (residual) vv += residual[idx];
                C[idx] = vv;
            }
        }
    }
}

// ---------------------------------------------------------------------------
// Per-(row,head) activations in place on qkvb: softmax(q), softmax(k),
// softmax(rk) over D=64; sigmoid(beta), sigmoid(rbeta). 1 wave per group.
// ---------------------------------------------------------------------------
__global__ __launch_bounds__(128)
void act_kernel(float* __restrict__ qkvb)
{
    int gidx = blockIdx.x * 4 + (threadIdx.x >> 5);
    int lane = threadIdx.x & 31;
    int row = gidx >> 3, h = gidx & 7;
    float* p = qkvb + (size_t)row * PROJ + h * 322;

    const int offs[3] = {0, 64, 192};   // q, k, rk
#pragma unroll
    for (int t = 0; t < 3; ++t) {
        float a = p[offs[t] + lane];
        float b = p[offs[t] + lane + 32];
        float m = fmaxf(a, b);
#pragma unroll
        for (int mm = 16; mm; mm >>= 1) m = fmaxf(m, __shfl_xor(m, mm, 32));
        float ea = __expf(a - m), eb = __expf(b - m);
        float s = ea + eb;
#pragma unroll
        for (int mm = 16; mm; mm >>= 1) s += __shfl_xor(s, mm, 32);
        float inv = 1.f / s;
        p[offs[t] + lane]      = ea * inv;
        p[offs[t] + lane + 32] = eb * inv;
    }
    if (lane < 2) {
        float bz = p[320 + lane];
        p[320 + lane] = 1.f / (1.f + __expf(-bz));  // sigmoid(beta), sigmoid(rbeta)
    }
}

// ---------------------------------------------------------------------------
// Fused sequential scans (delta-rule fast weight + fast RNN).
// One block per (b,h) chain; thread i owns row i of both 64x64 state
// matrices in registers (128 VGPRs). k/q/rk/softmax(h) broadcast via LDS.
// Writes hs directly as f16 for the output GEMM.
// ---------------------------------------------------------------------------
__global__ __launch_bounds__(64)
void scan_kernel(const float* __restrict__ qkvb, _Float16* __restrict__ hs)
{
    const int bh = blockIdx.x;          // b*NH + h
    const int b  = bh >> 3, h = bh & 7;
    const int i  = threadIdx.x;

    __shared__ float sk[64], sq[64], srk[64], sh[64], sqh[64];

    float Wr[64], Rr[64];
#pragma unroll
    for (int j = 0; j < 64; ++j) { Wr[j] = 0.f; Rr[j] = 0.f; }
    float hcur = 0.f;

    for (int t = 0; t < S_LEN; ++t) {
        size_t base = ((size_t)t * BSZ + b) * PROJ + (size_t)h * 322;
        float q_i  = qkvb[base + i];          // already softmaxed
        float k_i  = qkvb[base + 64 + i];
        float v_i  = qkvb[base + 128 + i];
        float rk_i = qkvb[base + 192 + i];
        float rv_i = qkvb[base + 256 + i];
        float bt   = qkvb[base + 320];        // sigmoid(beta)
        float rbt  = qkvb[base + 321];        // sigmoid(rbeta)

        __syncthreads();                      // prior iter done reading LDS
        sk[i] = k_i; sq[i] = q_i; srk[i] = rk_i; sh[i] = hcur;
        __syncthreads();

        // --- scan 1: delta-rule fast weight ---
        float vold = 0.f;
#pragma unroll
        for (int j = 0; j < 64; ++j) vold += Wr[j] * sk[j];
        float dv  = bt * (v_i - vold);
        float z_i = 0.f;
#pragma unroll
        for (int j = 0; j < 64; ++j) {
            float wn = Wr[j] + dv * sk[j];
            Wr[j] = wn;
            z_i += wn * sq[j];
        }

        // --- softmax(h_{t-1}) across the 64 threads ---
        float hmax = -3.0e38f;
#pragma unroll
        for (int j = 0; j < 64; ++j) hmax = fmaxf(hmax, sh[j]);
        float e_i = __expf(hcur - hmax);
        sqh[i] = e_i;
        __syncthreads();
        float esum = 0.f;
#pragma unroll
        for (int j = 0; j < 64; ++j) esum += sqh[j];
        float inv = 1.f / esum;

        // --- scan 2: recurrent fast weight ---
        float vold2 = 0.f;
#pragma unroll
        for (int j = 0; j < 64; ++j) vold2 += Rr[j] * srk[j];
        float dv2 = rbt * (rv_i - vold2);
        float hn  = z_i;
#pragma unroll
        for (int j = 0; j < 64; ++j) {
            float rn = Rr[j] + dv2 * srk[j];
            Rr[j] = rn;
            hn += rn * (sqh[j] * inv);
        }
        hcur = hn;
        hs[((size_t)t * BSZ + b) * (NH * HD) + h * HD + i] = (_Float16)hcur;
    }
}

// ---------------------------------------------------------------------------
// Launch
// ---------------------------------------------------------------------------
extern "C" void kernel_launch(void* const* d_in, const int* in_sizes, int n_in,
                              void* d_out, int out_size, void* d_ws, size_t ws_size,
                              hipStream_t stream)
{
    const float* x      = (const float*)d_in[0];   // (S,B,IN)
    const float* W_slow = (const float*)d_in[1];   // (PROJ,IN)
    const float* W_out  = (const float*)d_in[2];   // (IN, H*D)
    const float* gamma  = (const float*)d_in[3];
    const float* beta   = (const float*)d_in[4];
    float* out = (float*)d_out;

    char* ws = (char*)d_ws;
    size_t off = 0;
    auto alloc = [&](size_t bytes) -> void* {
        void* p = ws + off;
        off = (off + bytes + 255) & ~(size_t)255;
        return p;
    };
    _Float16* normed16 = (_Float16*)alloc((size_t)ROWS * IND * sizeof(_Float16));     // 4 MB
    _Float16* Wslow16  = (_Float16*)alloc((size_t)PROJ_PAD * IND * sizeof(_Float16)); // 2.7 MB
    _Float16* Wout16   = (_Float16*)alloc((size_t)IND * IND * sizeof(_Float16));      // 0.5 MB
    float*    qkvb     = (float*)   alloc((size_t)ROWS * PROJ * sizeof(float));       // 42 MB
    _Float16* hs16     = (_Float16*)alloc((size_t)ROWS * IND * sizeof(_Float16));     // 4 MB
    (void)ws_size;

    // 1) LayerNorm -> f16
    ln_kernel<<<ROWS / 4, 128, 0, stream>>>(x, gamma, beta, normed16, ROWS);

    // 2) Weights -> f16 (W_slow zero-padded to 2624 rows)
    {
        int n1 = PROJ_PAD * IND;
        cvt_f16_pad_kernel<<<(n1 + 255) / 256, 256, 0, stream>>>(W_slow, Wslow16,
                                                                 PROJ, IND, PROJ_PAD);
        int n2 = IND * IND;
        cvt_f16_pad_kernel<<<(n2 + 255) / 256, 256, 0, stream>>>(W_out, Wout16,
                                                                 IND, IND, IND);
    }

    // 3) Projection GEMM: qkvb = normed @ W_slow^T   (4096 x 2576 x 512)
    {
        dim3 grid(ROWS / 64, PROJ_PAD / 64);
        gemm_nt_wmma<<<grid, 128, 0, stream>>>(normed16, Wslow16, nullptr, qkvb,
                                               ROWS, PROJ, IND);
    }

    // 4) softmax / sigmoid in place
    act_kernel<<<(ROWS * NH) / 4, 128, 0, stream>>>(qkvb);

    // 5) fused dual sequential scan -> hs (f16)
    scan_kernel<<<BSZ * NH, 64, 0, stream>>>(qkvb, hs16);

    // 6) Output GEMM + residual: out = x + hs @ W_out^T   (4096 x 512 x 512)
    {
        dim3 grid(ROWS / 64, IND / 64);
        gemm_nt_wmma<<<grid, 128, 0, stream>>>(hs16, Wout16, x, out,
                                               ROWS, IND, IND);
    }
}
